// InvertedResidualBlock_6975026889024
// MI455X (gfx1250) — matrix operand order repaired and checked by reading, not compiled, not run on hardware
//
#include <hip/hip_runtime.h>
#include <math.h>

#define BATCH 16
#define INP   96
#define HID   576
#define OUP   96
#define IMGH  56
#define IMGW  56
#define HW    (IMGH*IMGW)        // 3136
#define KBUD  288                // round(0.5*576)
#define EPSV  1e-5f

// GEMM tiling: 128-thread blocks (4 waves), N-strip = 64 (3136 = 49*64 exact)
#define NSTRIP     64
#define KSTEP      32
#define LDS_ROW_B  144           // 64 bf16 = 128B data + 16B pad (bank spread)
#define LDS_ROW_U16 (LDS_ROW_B/2)

typedef __attribute__((ext_vector_type(16))) __bf16 v16bf;
typedef __attribute__((ext_vector_type(8)))  float  v8f;
typedef __attribute__((ext_vector_type(4)))  unsigned int u32x4;
typedef __attribute__((ext_vector_type(8)))  int  i32x8;
typedef __attribute__((ext_vector_type(4)))  int  i32x4;

union BFrag { unsigned int u[8]; v16bf v; };

#if __has_builtin(__builtin_amdgcn_tensor_load_to_lds) && __has_builtin(__builtin_amdgcn_s_wait_tensorcnt)
#define USE_TDM 1
#endif

__device__ __forceinline__ unsigned short f2bf(float f) {
    unsigned int u = __float_as_uint(f);
    unsigned int r = (u + 0x7fffu + ((u >> 16) & 1u)) >> 16;   // RNE
    return (unsigned short)r;
}
__device__ __forceinline__ float bf2f(unsigned short s) {
    return __uint_as_float(((unsigned int)s) << 16);
}

__device__ __forceinline__ unsigned lds_byte_offset(const void* p) {
    // generic -> LDS address space: numeric value is the byte offset in LDS
    return (unsigned)(unsigned long long)(__attribute__((address_space(3))) const char*)p;
}

// ---- stage one K=32 x N=64 bf16 tile of B into LDS (row stride LDS_ROW_B bytes) ----
#ifdef USE_TDM
__device__ __forceinline__ void tdm_stage_tile(const unsigned short* gsrc, unsigned lds_off) {
    unsigned long long ga = (unsigned long long)(size_t)gsrc;
    // D# group 0: count=1, lds_addr, global_addr, type=2
    u32x4 g0;
    g0.x = 1u;                                   // count=1, user mode
    g0.y = lds_off;                              // LDS byte address
    g0.z = (unsigned)(ga & 0xffffffffull);
    g0.w = (unsigned)((ga >> 32) & 0x01ffffffull) | (2u << 30);  // addr[56:32] | type=2
    // D# group 1
    // d0: wg_mask=0, data_size=1 (2B), pad_enable, pad_interval=4 (32 dwords=128B), pad_amount=3 (4 dwords=16B)
    unsigned d0 = (1u << 16) | (1u << 20) | (4u << 22) | (3u << 25);
    unsigned td0 = HW;                           // tensor_dim0
    unsigned td1 = 0x7fffffffu;                  // tensor_dim1 (no OOB in K by construction)
    unsigned d1 = (td0 & 0xffffu) << 16;                         // barrier addr=0 | dim0 lo
    unsigned d2 = ((td0 >> 16) & 0xffffu) | ((td1 & 0xffffu) << 16);
    unsigned d3 = ((td1 >> 16) & 0xffffu) | ((unsigned)NSTRIP << 16);   // tile_dim0=64
    unsigned d4 = (unsigned)KSTEP;                               // tile_dim1=32, tile_dim2=0
    unsigned d5 = (unsigned)HW;                                  // tensor_dim0_stride lo32
    unsigned d6 = 0;                                             // stride hi=0, dim1_stride lo=0
    unsigned d7 = 0;
    i32x8 g1 = { (int)d0, (int)d1, (int)d2, (int)d3, (int)d4, (int)d5, (int)d6, (int)d7 };
    i32x4 z4 = { 0, 0, 0, 0 };
    i32x8 z8 = { 0, 0, 0, 0, 0, 0, 0, 0 };
    __builtin_amdgcn_tensor_load_to_lds(g0, g1, z4, z4, z8, 0);
    __builtin_amdgcn_s_wait_tensorcnt(0);
}
#endif

__device__ __forceinline__ void manual_stage_tile(const unsigned short* gsrc,
                                                  unsigned short* lds, int tid,
                                                  long long row_stride_elems) {
    // 128 threads x 2 chunks of 16B : 64 rows x 4 chunks
    #pragma unroll
    for (int it = 0; it < 2; ++it) {
        int idx = tid + it * 128;
        int row = idx >> 2;
        int ch  = idx & 3;
        const u32x4* src = (const u32x4*)(gsrc + (long long)row * row_stride_elems + ch * 8);
        *(u32x4*)((char*)lds + row * LDS_ROW_B + ch * 16) = *src;
    }
}

// build B fragment for this wave from the staged LDS tile
__device__ __forceinline__ void build_bfrag(const unsigned short* tile, int wave,
                                            int lm, int lhi, BFrag& Bf) {
    const int ncol = wave * 16 + lm;
    const unsigned short* base = tile + ncol;      // element [k][ncol] at k*LDS_ROW_U16
    #pragma unroll
    for (int v = 0; v < 8; ++v) {
        int k = (v < 4) ? (lhi * 8 + 2 * v) : (16 + lhi * 8 + 2 * (v - 4));
        unsigned int lo = base[(k    ) * LDS_ROW_U16];
        unsigned int hi = base[(k + 1) * LDS_ROW_U16];
        Bf.u[v] = lo | (hi << 16);
    }
}

// ---------------- conversion kernels ----------------
__global__ void cvt_x_kernel(const float* __restrict__ x, unsigned short* __restrict__ xb, int n) {
    int i = blockIdx.x * blockDim.x + threadIdx.x;
    if (i < n) xb[i] = f2bf(x[i]);
}

__global__ void cvt_w_kernel(const float* __restrict__ w1, const float* __restrict__ w2,
                             unsigned short* __restrict__ w1b, unsigned short* __restrict__ w2b) {
    int i = blockIdx.x * blockDim.x + threadIdx.x;
    if (i < HID * INP) w1b[i] = f2bf(w1[i]);
    if (i < OUP * HID) w2b[i] = f2bf(w2[i]);
}

// ---------------- global average pool ----------------
__global__ void pool_kernel(const float* __restrict__ x, float* __restrict__ pooled) {
    __shared__ float red[256];
    int bc = blockIdx.x;
    const float* p = x + (size_t)bc * HW;
    float acc = 0.f;
    for (int i = threadIdx.x; i < HW; i += 256) acc += p[i];
    red[threadIdx.x] = acc;
    __syncthreads();
    for (int s = 128; s > 0; s >>= 1) {
        if (threadIdx.x < s) red[threadIdx.x] += red[threadIdx.x + s];
        __syncthreads();
    }
    if (threadIdx.x == 0) pooled[bc] = red[0] * (1.0f / (float)HW);
}

// ---------------- saliency FC + sigmoid + top-k binarize ----------------
__global__ void sal_kernel(const float* __restrict__ pooled, const float* __restrict__ w_sal,
                           const float* __restrict__ b_sal, float* __restrict__ vec) {
    __shared__ float s[HID];
    __shared__ float pl[INP];
    int b = blockIdx.x;
    for (int i = threadIdx.x; i < INP; i += 256) pl[i] = pooled[b * INP + i];
    __syncthreads();
    for (int o = threadIdx.x; o < HID; o += 256) {
        float acc = b_sal[o];
        const float* wr = w_sal + (size_t)o * INP;
        #pragma unroll 4
        for (int i = 0; i < INP; ++i) acc += pl[i] * wr[i];
        s[o] = 1.0f / (1.0f + __expf(-acc));
    }
    __syncthreads();
    for (int o = threadIdx.x; o < HID; o += 256) {
        float v = s[o];
        int cnt = 0;
        for (int j = 0; j < HID; ++j) cnt += (s[j] > v) ? 1 : 0;
        vec[b * HID + o] = (cnt < KBUD) ? 1.0f : 0.0f;   // cnt<k <=> s[o] >= kth largest
    }
}

// ---------------- spatial mask (1x1 conv -> threshold) ----------------
__global__ void mask_kernel(const float* __restrict__ x, const float* __restrict__ w_mask,
                            const float* __restrict__ b_mask, float* __restrict__ mask) {
    int i = blockIdx.x * blockDim.x + threadIdx.x;
    if (i >= BATCH * HW) return;
    int b = i / HW, p = i % HW;
    float acc = b_mask[0];
    const float* xp = x + (size_t)b * INP * HW + p;
    #pragma unroll 4
    for (int c = 0; c < INP; ++c) acc += xp[(size_t)c * HW] * w_mask[c];
    mask[i] = (acc >= 0.f) ? 1.0f : 0.0f;
}

// ---------------- 3x3 max-pool dilation ----------------
__global__ void dilate_kernel(const float* __restrict__ mask, float* __restrict__ maskd) {
    int i = blockIdx.x * blockDim.x + threadIdx.x;
    if (i >= BATCH * HW) return;
    int b = i / HW, p = i % HW;
    int y = p / IMGW, xc = p % IMGW;
    const float* mb = mask + (size_t)b * HW;
    float mx = 0.f;
    #pragma unroll
    for (int dy = -1; dy <= 1; ++dy) {
        int ny = y + dy; if (ny < 0 || ny >= IMGH) continue;
        #pragma unroll
        for (int dx = -1; dx <= 1; ++dx) {
            int nx = xc + dx; if (nx < 0 || nx >= IMGW) continue;
            mx = fmaxf(mx, mb[ny * IMGW + nx]);
        }
    }
    maskd[i] = mx;
}

// ---------------- GEMM1: h1 = relu(bn1(W1 @ xb)) * maskd * vector  (bf16 out) ----------------
__global__ void __launch_bounds__(128)
gemm1_kernel(const unsigned short* __restrict__ w1b,
             const unsigned short* __restrict__ xb,
             const float* __restrict__ g, const float* __restrict__ be,
             const float* __restrict__ mu, const float* __restrict__ va,
             const float* __restrict__ maskd, const float* __restrict__ vec,
             unsigned short* __restrict__ h1) {
    __shared__ __align__(16) unsigned short tile[KSTEP * LDS_ROW_U16];
    const int tid  = threadIdx.x;
    const int wave = tid >> 5;
    const int lane = tid & 31;
    const int lm   = lane & 15;
    const int lhi  = lane >> 4;
    const int b    = blockIdx.z;
    const int Mb   = blockIdx.y * 16;
    const int n0   = blockIdx.x * NSTRIP;
    const int n    = n0 + wave * 16 + lm;

    v8f acc = {};
    const unsigned short* Arow = w1b + (size_t)(Mb + lm) * INP;
    const unsigned short* Bt   = xb + (size_t)b * INP * HW + n0;

    #pragma unroll
    for (int k0 = 0; k0 < INP; k0 += KSTEP) {
        const unsigned short* gsrc = Bt + (size_t)k0 * HW;
#ifdef USE_TDM
        if (wave == 0) tdm_stage_tile(gsrc, lds_byte_offset(tile));
#else
        manual_stage_tile(gsrc, tile, tid, HW);
#endif
        __syncthreads();

        BFrag A, Bf;
        #pragma unroll
        for (int v = 0; v < 8; ++v) {
            int kA = k0 + ((v < 4) ? (2 * v + 8 * lhi) : (16 + 2 * (v - 4) + 8 * lhi));
            A.u[v] = *(const unsigned int*)(Arow + kA);
        }
        build_bfrag(tile, wave, lm, lhi, Bf);
        acc = __builtin_amdgcn_wmma_f32_16x16x32_bf16(
            false, A.v, false, Bf.v, (short)0, acc, false, false);
        __syncthreads();
    }

    const float md = maskd[b * HW + n];
    #pragma unroll
    for (int r = 0; r < 8; ++r) {
        int m = Mb + r + lhi * 8;
        float sc = g[m] * rsqrtf(va[m] + EPSV);
        float v  = acc[r] * sc + (be[m] - mu[m] * sc);
        v = fmaxf(v, 0.f) * md * vec[b * HID + m];
        h1[((size_t)b * HID + m) * HW + n] = f2bf(v);
    }
}

// ---------------- depthwise 3x3 + bn + relu + mask + vector (bf16 in/out) ----------------
__global__ void dw_kernel(const unsigned short* __restrict__ h1, const float* __restrict__ wdw,
                          const float* __restrict__ g, const float* __restrict__ be,
                          const float* __restrict__ mu, const float* __restrict__ va,
                          const float* __restrict__ mask, const float* __restrict__ vec,
                          unsigned short* __restrict__ h2) {
    long long i = (long long)blockIdx.x * blockDim.x + threadIdx.x;
    if (i >= (long long)BATCH * HID * HW) return;
    int p   = (int)(i % HW);
    int bc  = (int)(i / HW);
    int c   = bc % HID;
    int b   = bc / HID;
    int y = p / IMGW, xc = p % IMGW;
    const unsigned short* plane = h1 + (size_t)bc * HW;
    const float* w = wdw + c * 9;
    float acc = 0.f;
    #pragma unroll
    for (int dy = -1; dy <= 1; ++dy) {
        int ny = y + dy; if (ny < 0 || ny >= IMGH) continue;
        #pragma unroll
        for (int dx = -1; dx <= 1; ++dx) {
            int nx = xc + dx; if (nx < 0 || nx >= IMGW) continue;
            acc += bf2f(plane[ny * IMGW + nx]) * w[(dy + 1) * 3 + (dx + 1)];
        }
    }
    float sc = g[c] * rsqrtf(va[c] + EPSV);
    float v = acc * sc + (be[c] - mu[c] * sc);
    v = fmaxf(v, 0.f) * mask[b * HW + p] * vec[b * HID + c];
    h2[i] = f2bf(v);
}

// ---------------- GEMM2: out = x + bn2(W2 @ h2) * mask  (f32 out) ----------------
__global__ void __launch_bounds__(128)
gemm2_kernel(const unsigned short* __restrict__ w2b,
             const unsigned short* __restrict__ h2,
             const float* __restrict__ g, const float* __restrict__ be,
             const float* __restrict__ mu, const float* __restrict__ va,
             const float* __restrict__ mask, const float* __restrict__ x,
             float* __restrict__ out) {
    __shared__ __align__(16) unsigned short tile[KSTEP * LDS_ROW_U16];
    const int tid  = threadIdx.x;
    const int wave = tid >> 5;
    const int lane = tid & 31;
    const int lm   = lane & 15;
    const int lhi  = lane >> 4;
    const int b    = blockIdx.z;
    const int Mb   = blockIdx.y * 16;
    const int n0   = blockIdx.x * NSTRIP;
    const int n    = n0 + wave * 16 + lm;

    v8f acc = {};
    const unsigned short* Arow = w2b + (size_t)(Mb + lm) * HID;
    const unsigned short* Bt   = h2 + (size_t)b * HID * HW + n0;

    for (int k0 = 0; k0 < HID; k0 += KSTEP) {
        const unsigned short* gsrc = Bt + (size_t)k0 * HW;
#ifdef USE_TDM
        if (wave == 0) tdm_stage_tile(gsrc, lds_byte_offset(tile));
#else
        manual_stage_tile(gsrc, tile, tid, HW);
#endif
        __syncthreads();

        BFrag A, Bf;
        #pragma unroll
        for (int v = 0; v < 8; ++v) {
            int kA = k0 + ((v < 4) ? (2 * v + 8 * lhi) : (16 + 2 * (v - 4) + 8 * lhi));
            A.u[v] = *(const unsigned int*)(Arow + kA);
        }
        build_bfrag(tile, wave, lm, lhi, Bf);
        acc = __builtin_amdgcn_wmma_f32_16x16x32_bf16(
            false, A.v, false, Bf.v, (short)0, acc, false, false);
        __syncthreads();
    }

    const float mk = mask[b * HW + n];
    #pragma unroll
    for (int r = 0; r < 8; ++r) {
        int m = Mb + r + lhi * 8;
        float sc = g[m] * rsqrtf(va[m] + EPSV);
        float v  = acc[r] * sc + (be[m] - mu[m] * sc);
        size_t idx = ((size_t)b * OUP + m) * HW + n;
        out[idx] = x[idx] + v * mk;
    }
}

// ---------------- launch ----------------
extern "C" void kernel_launch(void* const* d_in, const int* in_sizes, int n_in,
                              void* d_out, int out_size, void* d_ws, size_t ws_size,
                              hipStream_t stream) {
    const float* x     = (const float*)d_in[0];
    const float* w_pw1 = (const float*)d_in[1];
    const float* bn1_g = (const float*)d_in[2];
    const float* bn1_b = (const float*)d_in[3];
    const float* bn1_m = (const float*)d_in[4];
    const float* bn1_v = (const float*)d_in[5];
    const float* w_dw  = (const float*)d_in[6];
    const float* bnd_g = (const float*)d_in[7];
    const float* bnd_b = (const float*)d_in[8];
    const float* bnd_m = (const float*)d_in[9];
    const float* bnd_v = (const float*)d_in[10];
    const float* w_pw2 = (const float*)d_in[11];
    const float* bn2_g = (const float*)d_in[12];
    const float* bn2_b = (const float*)d_in[13];
    const float* bn2_m = (const float*)d_in[14];
    const float* bn2_v = (const float*)d_in[15];
    const float* w_mask = (const float*)d_in[16];
    const float* b_mask = (const float*)d_in[17];
    const float* w_sal  = (const float*)d_in[18];
    const float* b_sal  = (const float*)d_in[19];
    float* out = (float*)d_out;

    char* ws = (char*)d_ws;
    size_t cur = 0;
    auto alloc = [&](size_t bytes) -> char* {
        char* p = ws + cur;
        cur = (cur + bytes + 255) & ~(size_t)255;
        return p;
    };
    unsigned short* xb   = (unsigned short*)alloc((size_t)BATCH * INP * HW * 2);
    unsigned short* h1   = (unsigned short*)alloc((size_t)BATCH * HID * HW * 2);
    unsigned short* h2   = (unsigned short*)alloc((size_t)BATCH * HID * HW * 2);
    float* mask   = (float*)alloc((size_t)BATCH * HW * 4);
    float* maskd  = (float*)alloc((size_t)BATCH * HW * 4);
    float* vec    = (float*)alloc((size_t)BATCH * HID * 4);
    float* pooled = (float*)alloc((size_t)BATCH * INP * 4);
    unsigned short* w1b = (unsigned short*)alloc((size_t)HID * INP * 2);
    unsigned short* w2b = (unsigned short*)alloc((size_t)OUP * HID * 2);
    (void)ws_size;

    const int nx = BATCH * INP * HW;
    cvt_x_kernel<<<(nx + 255) / 256, 256, 0, stream>>>(x, xb, nx);
    cvt_w_kernel<<<(HID * INP + 255) / 256, 256, 0, stream>>>(w_pw1, w_pw2, w1b, w2b);
    pool_kernel<<<BATCH * INP, 256, 0, stream>>>(x, pooled);
    sal_kernel<<<BATCH, 256, 0, stream>>>(pooled, w_sal, b_sal, vec);
    mask_kernel<<<(BATCH * HW + 255) / 256, 256, 0, stream>>>(x, w_mask, b_mask, mask);
    dilate_kernel<<<(BATCH * HW + 255) / 256, 256, 0, stream>>>(mask, maskd);

    dim3 g1(HW / NSTRIP, HID / 16, BATCH);   // (49, 36, 16)
    gemm1_kernel<<<g1, 128, 0, stream>>>(w1b, xb, bn1_g, bn1_b, bn1_m, bn1_v,
                                         maskd, vec, h1);

    long long ndw = (long long)BATCH * HID * HW;
    dw_kernel<<<(unsigned)((ndw + 255) / 256), 256, 0, stream>>>(
        h1, w_dw, bnd_g, bnd_b, bnd_m, bnd_v, mask, vec, h2);

    dim3 g2(HW / NSTRIP, OUP / 16, BATCH);   // (49, 6, 16)
    gemm2_kernel<<<g2, 128, 0, stream>>>(w2b, h2, bn2_g, bn2_b, bn2_m, bn2_v,
                                         mask, x, out);
}
